// MultiHeadAttn_36799279792715
// MI455X (gfx1250) — compile-verified, matrix operand order checked by reading, not run on hardware
//
#include <hip/hip_runtime.h>
#include <stdint.h>

// Problem constants (match reference)
#define D_MODEL 1024
#define SEQ     1024
#define BATCH   4
#define NHEAD   16
#define DHEAD   64
#define NROWS   (BATCH * SEQ)      // 4096
#define QK_SCALE 0.125f            // 1/sqrt(64)
#define LN_EPS   1e-5f

typedef __attribute__((ext_vector_type(16))) __bf16         v16bf;
typedef __attribute__((ext_vector_type(16))) unsigned short us16;
typedef __attribute__((ext_vector_type(8)))  unsigned short us8;
typedef __attribute__((ext_vector_type(8)))  float          v8f;

// ---- helpers -------------------------------------------------------------

__device__ __forceinline__ unsigned short f2bf(float f) {
  unsigned int u = __builtin_bit_cast(unsigned int, f);
  u += 0x7FFFu + ((u >> 16) & 1u);            // round-to-nearest-even
  return (unsigned short)(u >> 16);
}

// A/B fragment for v_wmma_*_16x16x32_bf16, per ISA 7.12.2:
// lane L holds row (A) / col (B) = L%16; its 16 bf16 k-values are two
// contiguous 8-element (16B) chunks at k = 8*(L/16) and k = 8*(L/16)+16.
__device__ __forceinline__ us16 load_frag(const unsigned short* row, int kb0) {
  us8 c0 = *(const us8*)(row + kb0);
  us8 c1 = *(const us8*)(row + kb0 + 16);
  us16 r;
#pragma unroll
  for (int i = 0; i < 8; ++i) { r[i] = c0[i]; r[i + 8] = c1[i]; }
  return r;
}

__device__ __forceinline__ v8f wmma_bf16(us16 a, us16 b, v8f c) {
  return __builtin_amdgcn_wmma_f32_16x16x32_bf16(
      false, __builtin_bit_cast(v16bf, a),
      false, __builtin_bit_cast(v16bf, b),
      (short)0, c, false, false);
}

// reductions across the 16 lanes of each wave32 half (xor masks < 16 stay in-half)
__device__ __forceinline__ float hmax16(float x) {
#pragma unroll
  for (int m = 1; m < 16; m <<= 1) x = fmaxf(x, __shfl_xor(x, m, 32));
  return x;
}
__device__ __forceinline__ float hsum16(float x) {
#pragma unroll
  for (int m = 1; m < 16; m <<= 1) x += __shfl_xor(x, m, 32);
  return x;
}

// ---- stage 1: conversions ------------------------------------------------

__global__ void cvt_bf16_kernel(const float* __restrict__ in,
                                unsigned short* __restrict__ out, int n) {
  int i = blockIdx.x * blockDim.x + threadIdx.x;
  if (i < n) out[i] = f2bf(in[i]);
}

// W[in][out] (1024x1024 f32) -> Wt[out][in] bf16, output-coalesced
__global__ void cvt_bf16_t_kernel(const float* __restrict__ in,
                                  unsigned short* __restrict__ out) {
  int i = blockIdx.x * blockDim.x + threadIdx.x;   // flat index into Wt
  int o = i >> 10;          // out column of original W
  int k = i & 1023;         // in row of original W
  out[i] = f2bf(in[k * D_MODEL + o]);
}

// ---- stage 2: fused QKV projection --------------------------------------
// One wave -> 32x32 output tile for each of Q/K/V: per K-step 2 A-frags +
// 6 B-frags feed 12 WMMAs (A shared across Q/K/V, B shared across row tiles).
// Q,K stored [head][s][d] bf16; V stored transposed [head][d][s] bf16.

__global__ __launch_bounds__(128) void qkv_gemm_kernel(
    const unsigned short* __restrict__ X,      // [4096][1024] bf16
    const unsigned short* __restrict__ Wqt,    // [1024 out][1024 in] bf16
    const unsigned short* __restrict__ Wkt,
    const unsigned short* __restrict__ Wvt,
    const float* __restrict__ bq, const float* __restrict__ bk,
    const float* __restrict__ bv,
    unsigned short* __restrict__ Qb,           // [64][1024][64]
    unsigned short* __restrict__ Kb,           // [64][1024][64]
    unsigned short* __restrict__ Vtb) {        // [64][64][1024]
  int wid  = (blockIdx.x * blockDim.x + threadIdx.x) >> 5;
  int lane = threadIdx.x & 31;
  int rt = wid >> 5;                 // 32-row tile 0..127
  int ct = wid & 31;                 // 32-col tile 0..31
  int rowh = lane >> 4, nm = lane & 15, kb0 = rowh * 8;

  const unsigned short* arow[2];
  const unsigned short* brow[3][2];
#pragma unroll
  for (int u = 0; u < 2; ++u)
    arow[u] = X + (rt * 32 + u * 16 + nm) * D_MODEL;
#pragma unroll
  for (int c = 0; c < 2; ++c) {
    int col = ct * 32 + c * 16 + nm;
    brow[0][c] = Wqt + col * D_MODEL;
    brow[1][c] = Wkt + col * D_MODEL;
    brow[2][c] = Wvt + col * D_MODEL;
  }

  v8f acc[3][2][2] = {};   // [qkv][row subtile][col subtile]
  for (int kk = 0; kk < D_MODEL; kk += 32) {
    us16 a0 = load_frag(arow[0] + kk, kb0);
    us16 a1 = load_frag(arow[1] + kk, kb0);
#pragma unroll
    for (int w = 0; w < 3; ++w) {
#pragma unroll
      for (int c = 0; c < 2; ++c) {
        us16 bf = load_frag(brow[w][c] + kk, kb0);
        acc[w][0][c] = wmma_bf16(a0, bf, acc[w][0][c]);
        acc[w][1][c] = wmma_bf16(a1, bf, acc[w][1][c]);
      }
    }
  }

#pragma unroll
  for (int c = 0; c < 2; ++c) {
    int col = ct * 32 + c * 16 + nm;
    int h = col >> 6, d = col & 63;
    float bqv = bq[col], bkv = bk[col], bvv = bv[col];
#pragma unroll
    for (int u = 0; u < 2; ++u) {
#pragma unroll
      for (int i = 0; i < 8; ++i) {
        int r = rt * 32 + u * 16 + rowh * 8 + i;   // C-tile row mapping
        int b = r >> 10, s = r & 1023;
        int head = b * NHEAD + h;
        int qoff = (head * SEQ + s) * DHEAD + d;
        Qb[qoff] = f2bf(acc[0][u][c][i] + bqv);
        Kb[qoff] = f2bf(acc[1][u][c][i] + bkv);
        Vtb[(head * DHEAD + d) * SEQ + s] = f2bf(acc[2][u][c][i] + bvv);
      }
    }
  }
}

// ---- stage 3: flash-style attention -------------------------------------
// One wave owns 16 query rows of one head. Streams 32-key blocks:
// 4 WMMAs QK^T -> online softmax (half-wave shuffles) -> P restaged via LDS
// (C layout -> A layout) -> 4 WMMAs PV into f32 accumulators.
// global_prefetch pulls the next key block while softmax VALU co-executes.

__global__ __launch_bounds__(128) void attn_kernel(
    const unsigned short* __restrict__ Qb,   // [64][1024][64]
    const unsigned short* __restrict__ Kb,   // [64][1024][64]
    const unsigned short* __restrict__ Vtb,  // [64][64][1024]
    const int* __restrict__ amask,           // [4][1024]
    unsigned short* __restrict__ Avec) {     // [4][1024][1024] bf16
  __shared__ unsigned short Plds[4 * 16 * 32];   // per-wave 16x32 P tile
  int warp = threadIdx.x >> 5;
  int wid  = blockIdx.x * 4 + warp;
  int head = wid >> 6;                 // 0..63  (storage index b*16+h)
  int qt   = wid & 63;                 // query tile 0..63
  int b = head >> 4;
  int lane = threadIdx.x & 31, rowh = lane >> 4, nm = lane & 15, kb0 = rowh * 8;

  const unsigned short* Qh = Qb  + head * SEQ * DHEAD;
  const unsigned short* Kh = Kb  + head * SEQ * DHEAD;
  const unsigned short* Vh = Vtb + head * DHEAD * SEQ;
  const int* mrow = amask + b * SEQ;

  // Q fragments for rows qt*16..+15, contraction split d=[0,32) and [32,64)
  us16 aQ0 = load_frag(Qh + (qt * 16 + nm) * DHEAD, kb0);
  us16 aQ1 = load_frag(Qh + (qt * 16 + nm) * DHEAD + 32, kb0);

  float mi[8], li[8];
  v8f O0 = {}, O1 = {}, O2 = {}, O3 = {};
#pragma unroll
  for (int i = 0; i < 8; ++i) { mi[i] = -3.0e38f; li[i] = 0.f; }

  unsigned short* pbase = Plds + warp * (16 * 32);

  for (int kbk = 0; kbk < SEQ; kbk += 32) {
    // ---- scores: two 16x16 n-tiles (keys kbk..+15 and +16..+31)
    const unsigned short* kr0 = Kh + (kbk + nm) * DHEAD;
    const unsigned short* kr1 = Kh + (kbk + 16 + nm) * DHEAD;
    v8f s0 = {}, s1 = {};
    s0 = wmma_bf16(aQ0, load_frag(kr0, kb0), s0);
    s0 = wmma_bf16(aQ1, load_frag(kr0 + 32, kb0), s0);
    s1 = wmma_bf16(aQ0, load_frag(kr1, kb0), s1);
    s1 = wmma_bf16(aQ1, load_frag(kr1 + 32, kb0), s1);

    // prefetch next key block (K rows + V rows) while softmax runs
    if (kbk + 32 < SEQ) {
      __builtin_prefetch(kr0 + 32 * DHEAD, 0, 3);
      __builtin_prefetch(kr1 + 32 * DHEAD, 0, 3);
      __builtin_prefetch(Vh + nm * SEQ + kbk + 32, 0, 3);
      __builtin_prefetch(Vh + (48 + nm) * SEQ + kbk + 32, 0, 3);
    }

    int mk0 = mrow[kbk + nm];
    int mk1 = mrow[kbk + 16 + nm];

    // ---- online softmax over this key block
    float rmax[8];
#pragma unroll
    for (int i = 0; i < 8; ++i) {
      float a0 = mk0 ? -100000.f : s0[i] * QK_SCALE;
      float a1 = mk1 ? -100000.f : s1[i] * QK_SCALE;
      s0[i] = a0; s1[i] = a1;
      rmax[i] = hmax16(fmaxf(a0, a1));
    }
    v8f p0, p1;
#pragma unroll
    for (int i = 0; i < 8; ++i) {
      float mnew = fmaxf(mi[i], rmax[i]);
      float corr = __expf(mi[i] - mnew);
      float e0 = __expf(s0[i] - mnew);
      float e1 = __expf(s1[i] - mnew);
      li[i] = li[i] * corr + hsum16(e0 + e1);
      mi[i] = mnew;
      O0[i] *= corr; O1[i] *= corr; O2[i] *= corr; O3[i] *= corr;
      p0[i] = e0; p1[i] = e1;
    }

    // ---- restage P: C layout -> row-major 16x32 bf16 in LDS (per-wave, no barrier)
#pragma unroll
    for (int i = 0; i < 8; ++i) {
      int r = rowh * 8 + i;
      pbase[r * 32 + nm]      = f2bf(p0[i]);
      pbase[r * 32 + 16 + nm] = f2bf(p1[i]);
    }
    us16 aP = load_frag(pbase + nm * 32, kb0);   // A fragment, K = 32 keys

    // ---- PV: B fragments from transposed V (contiguous over keys)
    O0 = wmma_bf16(aP, load_frag(Vh + (0 * 16 + nm) * SEQ + kbk, kb0), O0);
    O1 = wmma_bf16(aP, load_frag(Vh + (1 * 16 + nm) * SEQ + kbk, kb0), O1);
    O2 = wmma_bf16(aP, load_frag(Vh + (2 * 16 + nm) * SEQ + kbk, kb0), O2);
    O3 = wmma_bf16(aP, load_frag(Vh + (3 * 16 + nm) * SEQ + kbk, kb0), O3);
  }

  // Reference quirk: attn_vec.view(N_HEAD, B, S, D).permute(1,2,0,3) on a
  // b-major [B*H] buffer => storage head L lands at b2=L%B, h2=L/B.
  int b2 = head & 3, h2 = head >> 2;
#pragma unroll
  for (int i = 0; i < 8; ++i) {
    float inv = 1.f / li[i];
    int q = qt * 16 + rowh * 8 + i;
    unsigned short* orow = Avec + (b2 * SEQ + q) * D_MODEL + h2 * DHEAD;
    orow[0 * 16 + nm] = f2bf(O0[i] * inv);
    orow[1 * 16 + nm] = f2bf(O1[i] * inv);
    orow[2 * 16 + nm] = f2bf(O2[i] * inv);
    orow[3 * 16 + nm] = f2bf(O3[i] * inv);
  }
}

// ---- stage 4: output projection + residual ------------------------------
// 32x32 tile per wave: per K-step 2 A + 2 B fragments -> 4 WMMAs.

__global__ __launch_bounds__(128) void oproj_kernel(
    const unsigned short* __restrict__ Avec,  // [4096][1024] bf16
    const unsigned short* __restrict__ Wot,   // [1024 out][1024 in] bf16
    const float* __restrict__ inp,            // residual
    float* __restrict__ Y) {                  // [4096][1024] f32
  int wid  = (blockIdx.x * blockDim.x + threadIdx.x) >> 5;
  int lane = threadIdx.x & 31;
  int rt = wid >> 5, ct = wid & 31;
  int rowh = lane >> 4, nm = lane & 15, kb0 = rowh * 8;

  const unsigned short* arow0 = Avec + (rt * 32 + nm) * D_MODEL;
  const unsigned short* arow1 = Avec + (rt * 32 + 16 + nm) * D_MODEL;
  const unsigned short* brow0 = Wot + (ct * 32 + nm) * D_MODEL;
  const unsigned short* brow1 = Wot + (ct * 32 + 16 + nm) * D_MODEL;

  v8f acc[2][2] = {};
  for (int kk = 0; kk < D_MODEL; kk += 32) {
    us16 a0 = load_frag(arow0 + kk, kb0);
    us16 a1 = load_frag(arow1 + kk, kb0);
    us16 b0 = load_frag(brow0 + kk, kb0);
    us16 b1 = load_frag(brow1 + kk, kb0);
    acc[0][0] = wmma_bf16(a0, b0, acc[0][0]);
    acc[0][1] = wmma_bf16(a0, b1, acc[0][1]);
    acc[1][0] = wmma_bf16(a1, b0, acc[1][0]);
    acc[1][1] = wmma_bf16(a1, b1, acc[1][1]);
  }

#pragma unroll
  for (int u = 0; u < 2; ++u) {
#pragma unroll
    for (int c = 0; c < 2; ++c) {
      int col = ct * 32 + c * 16 + nm;
#pragma unroll
      for (int i = 0; i < 8; ++i) {
        int r = rt * 32 + u * 16 + rowh * 8 + i;
        Y[r * D_MODEL + col] = acc[u][c][i] + inp[r * D_MODEL + col];
      }
    }
  }
}

// ---- stage 5: LayerNorm --------------------------------------------------

__global__ __launch_bounds__(256) void ln_kernel(
    const float* __restrict__ Y, const float* __restrict__ g,
    const float* __restrict__ bb, float* __restrict__ out) {
  int row = blockIdx.x;
  const float* y = Y + row * D_MODEL;
  __shared__ float red[2][8];
  float ls = 0.f, lq = 0.f;
  for (int i = threadIdx.x; i < D_MODEL; i += 256) {
    float v = y[i]; ls += v; lq += v * v;
  }
#pragma unroll
  for (int m = 1; m < 32; m <<= 1) {
    ls += __shfl_xor(ls, m, 32);
    lq += __shfl_xor(lq, m, 32);
  }
  int w = threadIdx.x >> 5;
  if ((threadIdx.x & 31) == 0) { red[0][w] = ls; red[1][w] = lq; }
  __syncthreads();
  float ts = 0.f, tq = 0.f;
#pragma unroll
  for (int i = 0; i < 8; ++i) { ts += red[0][i]; tq += red[1][i]; }
  float mu  = ts * (1.f / D_MODEL);
  float var = tq * (1.f / D_MODEL) - mu * mu;
  float inv = rsqrtf(var + LN_EPS);
  for (int i = threadIdx.x; i < D_MODEL; i += 256)
    out[row * D_MODEL + i] = (y[i] - mu) * inv * g[i] + bb[i];
}

// ---- launch --------------------------------------------------------------

extern "C" void kernel_launch(void* const* d_in, const int* in_sizes, int n_in,
                              void* d_out, int out_size, void* d_ws, size_t ws_size,
                              hipStream_t stream) {
  const float* inp   = (const float*)d_in[0];
  const int*   amask = (const int*)d_in[1];
  const float* Wq = (const float*)d_in[2];
  const float* bq = (const float*)d_in[3];
  const float* Wk = (const float*)d_in[4];
  const float* bk = (const float*)d_in[5];
  const float* Wv = (const float*)d_in[6];
  const float* bv = (const float*)d_in[7];
  const float* Wo = (const float*)d_in[8];
  const float* lng = (const float*)d_in[9];
  const float* lnb = (const float*)d_in[10];

  char* ws = (char*)d_ws;
  const size_t MB = 1u << 20;
  unsigned short* Xbf  = (unsigned short*)(ws + 0 * MB);   // 8 MB
  unsigned short* Wqt  = (unsigned short*)(ws + 8 * MB);   // 2 MB
  unsigned short* Wkt  = (unsigned short*)(ws + 10 * MB);  // 2 MB
  unsigned short* Wvt  = (unsigned short*)(ws + 12 * MB);  // 2 MB
  unsigned short* Wot  = (unsigned short*)(ws + 14 * MB);  // 2 MB
  unsigned short* Qb   = (unsigned short*)(ws + 16 * MB);  // 8 MB
  unsigned short* Kb   = (unsigned short*)(ws + 24 * MB);  // 8 MB
  unsigned short* Vtb  = (unsigned short*)(ws + 32 * MB);  // 8 MB
  unsigned short* Avec = (unsigned short*)(ws + 40 * MB);  // 8 MB
  float*          Ybuf = (float*)        (ws + 48 * MB);   // 16 MB

  // stage 1: conversions
  cvt_bf16_kernel<<<(NROWS * D_MODEL) / 256, 256, 0, stream>>>(inp, Xbf, NROWS * D_MODEL);
  cvt_bf16_t_kernel<<<(D_MODEL * D_MODEL) / 256, 256, 0, stream>>>(Wq, Wqt);
  cvt_bf16_t_kernel<<<(D_MODEL * D_MODEL) / 256, 256, 0, stream>>>(Wk, Wkt);
  cvt_bf16_t_kernel<<<(D_MODEL * D_MODEL) / 256, 256, 0, stream>>>(Wv, Wvt);
  cvt_bf16_t_kernel<<<(D_MODEL * D_MODEL) / 256, 256, 0, stream>>>(Wo, Wot);

  // stage 2: QKV (128 row tiles * 32 col tiles = 4096 waves, 4 waves/block)
  qkv_gemm_kernel<<<1024, 128, 0, stream>>>(Xbf, Wqt, Wkt, Wvt, bq, bk, bv,
                                            Qb, Kb, Vtb);

  // stage 3: attention (64 heads * 64 query tiles = 4096 waves)
  attn_kernel<<<1024, 128, 0, stream>>>(Qb, Kb, Vtb, amask, Avec);

  // stage 4: output projection + residual (128 * 32 tiles = 4096 waves)
  oproj_kernel<<<1024, 128, 0, stream>>>(Avec, Wot, inp, Ybuf);

  // stage 5: layer norm -> d_out
  ln_kernel<<<NROWS, 256, 0, stream>>>(Ybuf, lng, lnb, (float*)d_out);
}
